// RetrievalTool_42580305772657
// MI455X (gfx1250) — matrix operand order repaired and checked by reading, not compiled, not run on hardware
//
#include <hip/hip_runtime.h>
#include <hip/hip_bf16.h>
#include <math.h>

typedef __attribute__((ext_vector_type(16))) _Float16 v16h;
typedef __attribute__((ext_vector_type(4)))  _Float16 v4h;
typedef __attribute__((ext_vector_type(8)))  float    v8f;

// types for the gfx1250 async global->LDS builtin (signature revealed by the
// compiler: param 1 is "__vector_size__(16) int __device__ *", i.e. AS1 v4i*)
typedef int v4i_ __attribute__((__vector_size__(16)));
typedef __attribute__((address_space(1))) v4i_ gv4i_t;
typedef __attribute__((address_space(3))) v4i_ sv4i_t;

#define B_      64
#define N_      20000
#define L_      336
#define P_      96
#define C_      16
#define D_      64
#define LC      (L_*C_)          /* 5376 */
#define LC4     (LC/4)           /* 1344 */
#define PC      (P_*C_)          /* 1536 */
#define PC4     (PC/4)           /* 384  */
#define N4      (N_/4)           /* 5000 */
#define KCOARSE 80
#define TOPM    20
#define N_TAB   17
#define CAT_DIM 16
#define CATIN   (N_TAB*CAT_DIM)  /* 272 */
#define GH      128
#define GIN     (2*D_+2)         /* 130 */

// CDNA5 async global->LDS copy path (ASYNCcnt-tracked), guarded so the file
// compiles on toolchains without the builtins (and on the host pass).
#if defined(__AMDGCN__) && __has_builtin(__builtin_amdgcn_global_load_async_to_lds_b128) && __has_builtin(__builtin_amdgcn_s_wait_asynccnt)
#define ASYNC_LDS 1
#else
#define ASYNC_LDS 0
#endif

__device__ __constant__ int c_idmin[N_TAB] = {0,0,0,0,0,1,1,0,0,0,0,0,0,0,0,0,0};
__device__ __constant__ int c_idmax[N_TAB] = {2047,255,4095,23,6,12,53,3,1,1,15,7,2,2,2,2,1};

__device__ __forceinline__ float gelu_exact(float x) {
    return 0.5f * x * (1.0f + erff(x * 0.70710678118654752440f));
}

__device__ __forceinline__ float blockReduceSum64(float v, float* red, int tid) {
    red[tid] = v; __syncthreads();
    for (int s = 32; s > 0; s >>= 1) {
        if (tid < s) red[tid] += red[tid + s];
        __syncthreads();
    }
    float r = red[0]; __syncthreads();
    return r;
}

// ---------------- 1. context encoder: q[b,:] (fp32) and unit(q) (f16) ----------
__global__ void encoder_kernel(const int* __restrict__ meta_ids,
                               const float* __restrict__ exo,
                               const float* __restrict__ ls,
                               const float* __restrict__ emb,
                               const float* __restrict__ cw1, const float* __restrict__ cb1,
                               const float* __restrict__ cw2, const float* __restrict__ cb2,
                               const float* __restrict__ ew1, const float* __restrict__ eb1,
                               const float* __restrict__ ew2, const float* __restrict__ eb2,
                               const float* __restrict__ lw1, const float* __restrict__ lb1,
                               const float* __restrict__ lw2, const float* __restrict__ lb2,
                               const float* __restrict__ lng, const float* __restrict__ lnb,
                               float* __restrict__ q_out, _Float16* __restrict__ qh_out) {
    __shared__ float cat[CATIN];
    __shared__ float h[D_];
    __shared__ float red[D_];
    const int b = blockIdx.x, d = threadIdx.x;

    for (int j = d; j < CATIN; j += D_) {
        int t = j / CAT_DIM, c = j % CAT_DIM;
        int id = meta_ids[b * N_TAB + t];
        id = min(max(id, c_idmin[t]), c_idmax[t]);
        cat[j] = emb[((size_t)t * 4096 + (size_t)id) * CAT_DIM + c];
    }
    __syncthreads();

    float acc = cb1[d];
    for (int j = 0; j < CATIN; ++j) acc += cat[j] * cw1[j * D_ + d];
    h[d] = gelu_exact(acc);
    __syncthreads();
    float ccd = cb2[d];
    for (int j = 0; j < D_; ++j) ccd += h[j] * cw2[j * D_ + d];
    __syncthreads();

    acc = eb1[d];
    for (int j = 0; j < 8; ++j) acc += exo[b * 8 + j] * ew1[j * D_ + d];
    h[d] = gelu_exact(acc);
    __syncthreads();
    float e2 = eb2[d];
    for (int j = 0; j < D_; ++j) e2 += h[j] * ew2[j * D_ + d];
    ccd += e2;
    __syncthreads();

    float qd = 0.0f;
    for (int r = 0; r < 3; ++r) {
        acc = lb1[d];
        for (int j = 0; j < 4; ++j) acc += ls[(b * 3 + r) * 4 + j] * lw1[j * D_ + d];
        h[d] = gelu_exact(acc);
        __syncthreads();
        float td = ccd + lb2[d];
        for (int j = 0; j < D_; ++j) td += h[j] * lw2[j * D_ + d];
        float m = blockReduceSum64(td, red, d) * (1.0f / 64.0f);
        float dev = td - m;
        float v = blockReduceSum64(dev * dev, red, d) * (1.0f / 64.0f);
        qd += dev * rsqrtf(v + 1e-5f) * lng[d] + lnb[d];
        __syncthreads();
    }
    qd *= (1.0f / 3.0f);
    q_out[b * D_ + d] = qd;
    float nrm = sqrtf(blockReduceSum64(qd * qd, red, d));
    qh_out[b * D_ + d] = (_Float16)(qd / (nrm + 1e-12f));
}

// ---------------- 2. normalize pool_context rows -> f16 ------------------------
__global__ void pcn_kernel(const float* __restrict__ pc, _Float16* __restrict__ pcn) {
    int n = blockIdx.x * blockDim.x + threadIdx.x;
    if (n >= N_) return;
    const float4* r4 = (const float4*)(pc + (size_t)n * D_);
    float rowv[D_];
    float ss = 0.0f;
    for (int j = 0; j < D_ / 4; ++j) {
        float4 v = r4[j];
        rowv[4 * j + 0] = v.x; rowv[4 * j + 1] = v.y;
        rowv[4 * j + 2] = v.z; rowv[4 * j + 3] = v.w;
        ss += v.x * v.x + v.y * v.y + v.z * v.z + v.w * v.w;
    }
    float inv = 1.0f / (sqrtf(ss) + 1e-12f);
    v4h* dst = (v4h*)(pcn + (size_t)n * D_);
    for (int j = 0; j < D_ / 4; ++j) {
        v4h o;
        o[0] = (_Float16)(rowv[4 * j + 0] * inv);
        o[1] = (_Float16)(rowv[4 * j + 1] * inv);
        o[2] = (_Float16)(rowv[4 * j + 2] * inv);
        o[3] = (_Float16)(rowv[4 * j + 3] * inv);
        dst[j] = o;
    }
}

// ---------------- 3. ||x_b|| ---------------------------------------------------
__global__ void xnorm_kernel(const float* __restrict__ x, float* __restrict__ xn) {
    __shared__ float red[256];
    const int b = blockIdx.x, tid = threadIdx.x;
    const float4* r4 = (const float4*)(x + (size_t)b * LC);
    float ss = 0.0f;
    for (int i = tid; i < LC4; i += 256) {
        float4 v = r4[i];
        ss += v.x * v.x + v.y * v.y + v.z * v.z + v.w * v.w;
    }
    red[tid] = ss; __syncthreads();
    for (int s = 128; s > 0; s >>= 1) { if (tid < s) red[tid] += red[tid + s]; __syncthreads(); }
    if (tid == 0) xn[b] = sqrtf(red[0]);
}

// ---------------- 4. meta_sim = unit(q) @ unit(pool_context)^T via WMMA --------
__global__ void sim_wmma_kernel(const _Float16* __restrict__ qh,
                                const _Float16* __restrict__ pcn,
                                float* __restrict__ sim) {
    const int nbase = blockIdx.x * 16;
    const int wave  = threadIdx.x >> 5;
    const int lane  = threadIdx.x & 31;
    const int mbase = wave * 16;
    const int mrow  = mbase + (lane & 15);
    const int ncol  = nbase + (lane & 15);
    const int kbA   = (lane >> 4) * 16;   // A: lanes 0-15 -> K 0-15, 16-31 -> K 16-31
    const int kbB   = (lane >> 4) * 16;   // B: lanes 0-15 -> K 0-15, 16-31 -> K 16-31

    v16h a0, a1, b0, b1;
    const _Float16* qrow = qh + (size_t)mrow * D_;
    const _Float16* prow = pcn + (size_t)ncol * D_;
#pragma unroll
    for (int i = 0; i < 8; ++i) {
        a0[i]     = qrow[kbA + i];
        a0[i + 8] = qrow[kbA + 8 + i];
        a1[i]     = qrow[32 + kbA + i];
        a1[i + 8] = qrow[32 + kbA + 8 + i];
    }
#pragma unroll
    for (int i = 0; i < 16; ++i) {
        b0[i] = prow[kbB + i];
        b1[i] = prow[32 + kbB + i];
    }
    v8f c = {};
    c = __builtin_amdgcn_wmma_f32_16x16x32_f16(false, a0, false, b0, (short)0, c, false, false);
    c = __builtin_amdgcn_wmma_f32_16x16x32_f16(false, a1, false, b1, (short)0, c, false, false);

    const int mhi = (lane >> 4) * 8;
#pragma unroll
    for (int r = 0; r < 8; ++r) {
        int m = mbase + r + mhi;
        sim[(size_t)m * N_ + ncol] = c[r];
    }
}

// ---------------- 5. top-80 per row: whole row in LDS (80KB of 320KB WGP LDS) --
__global__ void topk_kernel(const float* __restrict__ sim,
                            int* __restrict__ cidx, float* __restrict__ metc) {
    extern __shared__ float smem[];
    float* vals = smem;                 // N_ floats
    float* rv   = smem + N_;            // 512 floats
    int*   ri   = (int*)(rv + 512);     // 512 ints
    const int b = blockIdx.x, tid = threadIdx.x;
    const float4* row4 = (const float4*)(sim + (size_t)b * N_);
    float4* vals4 = (float4*)vals;
    for (int i = tid; i < N4; i += 512) vals4[i] = row4[i];
    __syncthreads();
    for (int k = 0; k < KCOARSE; ++k) {
        float bm = -3.0e38f; int bi = N_;
        for (int i = tid; i < N_; i += 512) {
            float v = vals[i];
            if (v > bm) { bm = v; bi = i; }
        }
        rv[tid] = bm; ri[tid] = bi; __syncthreads();
        for (int s = 256; s > 0; s >>= 1) {
            if (tid < s) {
                float ov = rv[tid + s]; int oi = ri[tid + s];
                if (ov > rv[tid] || (ov == rv[tid] && oi < ri[tid])) { rv[tid] = ov; ri[tid] = oi; }
            }
            __syncthreads();
        }
        if (tid == 0) {
            cidx[b * KCOARSE + k] = ri[0];
            metc[b * KCOARSE + k] = rv[0];
            vals[ri[0]] = -3.0e38f;
        }
        __syncthreads();
    }
}

// ---------------- 6. fine series similarity + combine --------------------------
// Dominant HBM stream (~110MB gathered pool_x): float4 everywhere; x row staged
// into LDS via CDNA5 async copy when available; prefetch the pool row meanwhile.
__global__ void fine_kernel(const float* __restrict__ x, const float* __restrict__ pool_x,
                            const int* __restrict__ cidx, const float* __restrict__ metc,
                            const float* __restrict__ xn, float* __restrict__ comb) {
    __shared__ float xs[LC];
    __shared__ float redD[256], redS[256];
    const int k = blockIdx.x, b = blockIdx.y, tid = threadIdx.x;
    const int idx = cidx[b * KCOARSE + k];
    const float4* xr4 = (const float4*)(x + (size_t)b * LC);
    const float4* pr4 = (const float4*)(pool_x + (size_t)idx * LC);
    float4* xs4 = (float4*)xs;

    // warm the gathered pool row while we stage x (global_prefetch_b8)
    __builtin_prefetch((const void*)(pr4 + tid), 0, 1);

#if ASYNC_LDS
    for (int i = tid; i < LC4; i += 256) {
        __builtin_amdgcn_global_load_async_to_lds_b128(
            (gv4i_t*)(unsigned long long)(const void*)(xr4 + i),
            (sv4i_t*)(unsigned)(unsigned long long)(const void*)(xs4 + i),
            0, 0);
    }
    __builtin_amdgcn_s_wait_asynccnt(0);
#else
    for (int i = tid; i < LC4; i += 256) xs4[i] = xr4[i];
#endif
    __syncthreads();

    float dot = 0.0f, ss = 0.0f;
    for (int i = tid; i < LC4; i += 256) {
        float4 v = pr4[i];
        float4 xv = xs4[i];
        dot += xv.x * v.x + xv.y * v.y + xv.z * v.z + xv.w * v.w;
        ss  += v.x * v.x + v.y * v.y + v.z * v.z + v.w * v.w;
    }
    redD[tid] = dot; redS[tid] = ss; __syncthreads();
    for (int s = 128; s > 0; s >>= 1) {
        if (tid < s) { redD[tid] += redD[tid + s]; redS[tid] += redS[tid + s]; }
        __syncthreads();
    }
    if (tid == 0) {
        float series = redD[0] / ((xn[b] + 1e-12f) * (sqrtf(redS[0]) + 1e-12f));
        comb[b * KCOARSE + k] = 0.7f * metc[b * KCOARSE + k] + 0.3f * series;
    }
}

// ---------------- 7. top-20 + gate MLP + softmax + weighted pool_y -------------
__global__ void final_kernel(const float* __restrict__ comb, const float* __restrict__ metc,
                             const int* __restrict__ cidx, const float* __restrict__ q,
                             const float* __restrict__ pc, const float* __restrict__ pool_y,
                             const float* __restrict__ gw1, const float* __restrict__ gb1,
                             const float* __restrict__ gw2, const float* __restrict__ gb2,
                             float* __restrict__ out) {
    __shared__ float cv[KCOARSE];
    __shared__ float sel_c[TOPM], sel_m[TOPM];
    __shared__ int   sel_i[TOPM];
    __shared__ float gi[GIN];
    __shared__ float red[GH];
    __shared__ float wgt[TOPM];
    const int b = blockIdx.x, tid = threadIdx.x;   // GH = 128 threads

    for (int i = tid; i < KCOARSE; i += GH) cv[i] = comb[b * KCOARSE + i];
    __syncthreads();
    if (tid == 0) {
        for (int m = 0; m < TOPM; ++m) {
            float bm = -3.0e38f; int bi = 0;
            for (int i = 0; i < KCOARSE; ++i) if (cv[i] > bm) { bm = cv[i]; bi = i; }
            sel_c[m] = bm;
            sel_m[m] = metc[b * KCOARSE + bi];
            sel_i[m] = cidx[b * KCOARSE + bi];
            cv[bi] = -3.0e38f;
        }
    }
    __syncthreads();

    for (int m = 0; m < TOPM; ++m) {
        if (tid < D_)      gi[tid] = q[b * D_ + tid];
        else               gi[tid] = pc[(size_t)sel_i[m] * D_ + (tid - D_)];
        if (tid == 0) { gi[2 * D_] = sel_c[m]; gi[2 * D_ + 1] = sel_m[m]; }
        __syncthreads();
        float acc = gb1[tid];
        for (int j = 0; j < GIN; ++j) acc += gi[j] * gw1[j * GH + tid];
        red[tid] = gelu_exact(acc) * gw2[tid];
        __syncthreads();
        for (int s = 64; s > 0; s >>= 1) { if (tid < s) red[tid] += red[tid + s]; __syncthreads(); }
        if (tid == 0) wgt[m] = sel_c[m] * 10.0f + red[0] + gb2[0];   // comb/TEMP + gate
        __syncthreads();
    }

    if (tid == 0) {
        float mx = -3.0e38f;
        for (int m = 0; m < TOPM; ++m) mx = fmaxf(mx, wgt[m]);
        float s = 0.0f;
        for (int m = 0; m < TOPM; ++m) { wgt[m] = expf(wgt[m] - mx); s += wgt[m]; }
        float inv = 1.0f / s;
        for (int m = 0; m < TOPM; ++m) wgt[m] *= inv;
    }
    __syncthreads();

    // PC4 = 384 = 3 * GH : exactly 3 float4 per thread
    float4* out4 = (float4*)(out + (size_t)b * PC);
    for (int e = tid; e < PC4; e += GH) {
        float4 acc = make_float4(0.f, 0.f, 0.f, 0.f);
        for (int m = 0; m < TOPM; ++m) {
            const float4* yr4 = (const float4*)(pool_y + (size_t)sel_i[m] * PC);
            float4 y = yr4[e];
            float w = wgt[m];
            acc.x += w * y.x; acc.y += w * y.y; acc.z += w * y.z; acc.w += w * y.w;
        }
        out4[e] = acc;
    }
}

// ---------------- workspace layout ---------------------------------------------
#define ALIGN256(x) (((x) + (size_t)255) & ~(size_t)255)
static constexpr size_t OFF_Q    = 0;
static constexpr size_t OFF_QH   = ALIGN256(OFF_Q    + (size_t)B_ * D_ * 4);
static constexpr size_t OFF_PCN  = ALIGN256(OFF_QH   + (size_t)B_ * D_ * 2);
static constexpr size_t OFF_SIM  = ALIGN256(OFF_PCN  + (size_t)N_ * D_ * 2);
static constexpr size_t OFF_XN   = ALIGN256(OFF_SIM  + (size_t)B_ * N_ * 4);
static constexpr size_t OFF_CIDX = ALIGN256(OFF_XN   + (size_t)B_ * 4);
static constexpr size_t OFF_METC = ALIGN256(OFF_CIDX + (size_t)B_ * KCOARSE * 4);
static constexpr size_t OFF_COMB = ALIGN256(OFF_METC + (size_t)B_ * KCOARSE * 4);

extern "C" void kernel_launch(void* const* d_in, const int* in_sizes, int n_in,
                              void* d_out, int out_size, void* d_ws, size_t ws_size,
                              hipStream_t stream) {
    const int*   meta_ids = (const int*)  d_in[0];
    const float* exo      = (const float*)d_in[1];
    const float* ls       = (const float*)d_in[2];
    const float* x        = (const float*)d_in[3];
    const float* pc       = (const float*)d_in[4];
    const float* pool_x   = (const float*)d_in[5];
    const float* pool_y   = (const float*)d_in[6];
    const float* emb      = (const float*)d_in[7];
    const float* cw1 = (const float*)d_in[8];  const float* cb1 = (const float*)d_in[9];
    const float* cw2 = (const float*)d_in[10]; const float* cb2 = (const float*)d_in[11];
    const float* ew1 = (const float*)d_in[12]; const float* eb1 = (const float*)d_in[13];
    const float* ew2 = (const float*)d_in[14]; const float* eb2 = (const float*)d_in[15];
    const float* lw1 = (const float*)d_in[16]; const float* lb1 = (const float*)d_in[17];
    const float* lw2 = (const float*)d_in[18]; const float* lb2 = (const float*)d_in[19];
    const float* lng = (const float*)d_in[20]; const float* lnb = (const float*)d_in[21];
    const float* gw1 = (const float*)d_in[22]; const float* gb1 = (const float*)d_in[23];
    const float* gw2 = (const float*)d_in[24]; const float* gb2 = (const float*)d_in[25];
    float* out = (float*)d_out;

    char* ws = (char*)d_ws;
    float*    q_ws   = (float*)   (ws + OFF_Q);
    _Float16* qh_ws  = (_Float16*)(ws + OFF_QH);
    _Float16* pcn_ws = (_Float16*)(ws + OFF_PCN);
    float*    sim_ws = (float*)   (ws + OFF_SIM);
    float*    xn_ws  = (float*)   (ws + OFF_XN);
    int*      cidx   = (int*)     (ws + OFF_CIDX);
    float*    metc   = (float*)   (ws + OFF_METC);
    float*    comb   = (float*)   (ws + OFF_COMB);

    encoder_kernel<<<B_, D_, 0, stream>>>(meta_ids, exo, ls, emb,
        cw1, cb1, cw2, cb2, ew1, eb1, ew2, eb2, lw1, lb1, lw2, lb2,
        lng, lnb, q_ws, qh_ws);

    pcn_kernel<<<(N_ + 255) / 256, 256, 0, stream>>>(pc, pcn_ws);
    xnorm_kernel<<<B_, 256, 0, stream>>>(x, xn_ws);

    sim_wmma_kernel<<<N_ / 16, 128, 0, stream>>>(qh_ws, pcn_ws, sim_ws);

    const size_t topk_lds = (size_t)N_ * 4 + 512 * 4 + 512 * 4;   // 84 KB < 320 KB WGP LDS
    topk_kernel<<<B_, 512, topk_lds, stream>>>(sim_ws, cidx, metc);

    fine_kernel<<<dim3(KCOARSE, B_), 256, 0, stream>>>(x, pool_x, cidx, metc, xn_ws, comb);

    final_kernel<<<B_, GH, 0, stream>>>(comb, metc, cidx, q_ws, pc, pool_y,
                                        gw1, gb1, gw2, gb2, out);
}
